// Physics_Attention_Structured_Mesh_2D_20933670601310
// MI455X (gfx1250) — matrix operand order repaired and checked by reading, not compiled, not run on hardware
//
#include <hip/hip_runtime.h>
#include <cmath>

typedef __bf16 bf16;
typedef __bf16 v16bf __attribute__((ext_vector_type(16)));
typedef float  v8f   __attribute__((ext_vector_type(8)));

#define DEV __device__ __forceinline__

namespace cfg {
constexpr int B = 2, Hh = 160, Ww = 160, NT = 160 * 160;   // 25600 tokens
constexpr int DIM = 256, HEADS = 8, DH = 64, INNER = 512, G = 64;
constexpr int NBH = B * HEADS;                             // 16
constexpr int KC = 9 * DIM;                                // conv GEMM K = 2304
}

struct __attribute__((aligned(16))) bf16x8 { bf16 v[8]; };

// Build a 16-element bf16 WMMA fragment from two contiguous 16-byte chunks.
// A-frag: lo = row + kb + half*8, hi = row + kb + 16 + half*8 (vgprs 0-3 / 4-7)
// B-frag: lo = col + kb + half*8, hi = col + kb + 16 + half*8 (same k order as A)
DEV v16bf frag16(const bf16* lo, const bf16* hi) {
  bf16x8 a = *(const bf16x8*)lo;
  bf16x8 b = *(const bf16x8*)hi;
  v16bf r;
#pragma unroll
  for (int j = 0; j < 8; ++j) { r[j] = a.v[j]; r[8 + j] = b.v[j]; }
  return r;
}

// ---------------- conversion / layout kernels --------------------------------
__global__ void cvt_f32_bf16(const float* __restrict__ x, bf16* __restrict__ y, size_t n) {
  for (size_t i = blockIdx.x * (size_t)blockDim.x + threadIdx.x; i < n;
       i += (size_t)gridDim.x * blockDim.x)
    y[i] = (bf16)x[i];
}

// conv_w [O=512][Cin=256][3][3] -> wBT[o][k=(ky*3+kx)*256+cin]  (K contiguous per column)
__global__ void cvt_convw(const float* __restrict__ w, bf16* __restrict__ wBT) {
  using namespace cfg;
  size_t n = (size_t)KC * INNER;
  for (size_t i = blockIdx.x * (size_t)blockDim.x + threadIdx.x; i < n;
       i += (size_t)gridDim.x * blockDim.x) {
    int o = (int)(i / KC), k = (int)(i % KC);
    int plane = k >> 8, cin = k & 255;
    int ky = plane / 3, kx = plane % 3;
    wBT[i] = (bf16)w[(((size_t)o * DIM + cin) * 3 + ky) * 3 + kx];
  }
}

// xp [B][NT][512] -> xpT [bh][c][n] bf16 (tiled transpose through LDS, coalesced both ways)
__global__ void __launch_bounds__(256)
transpose_xproj(const float* __restrict__ xp, bf16* __restrict__ xpT) {
  using namespace cfg;
  int tile = blockIdx.x % (NT / 64);
  int bh = blockIdx.x / (NT / 64);
  int b = bh >> 3, h = bh & 7;
  int n0 = tile * 64;
  __shared__ bf16 T[64][65];
  for (int e = threadIdx.x; e < 64 * 64; e += 256) {
    int r = e >> 6, c = e & 63;   // token r (coalesced over c)
    T[r][c] = (bf16)xp[((size_t)b * NT + n0 + r) * INNER + h * DH + c];
  }
  __syncthreads();
  for (int e = threadIdx.x; e < 64 * 64; e += 256) {
    int c = e >> 6, r = e & 63;   // coalesced over r
    xpT[((size_t)bh * DH + c) * NT + n0 + r] = T[r][c];
  }
}

// wT [bh][g][n] -> wN [b][n][h*64+g] (A-operand layout for the final GEMM)
__global__ void __launch_bounds__(256)
transpose_wn(const bf16* __restrict__ wT, bf16* __restrict__ wN) {
  using namespace cfg;
  int tile = blockIdx.x % (NT / 64);
  int bh = blockIdx.x / (NT / 64);
  int b = bh >> 3, h = bh & 7;
  int n0 = tile * 64;
  __shared__ bf16 T[64][65];
  for (int e = threadIdx.x; e < 64 * 64; e += 256) {
    int g = e >> 6, n = e & 63;
    T[g][n] = wT[((size_t)bh * G + g) * NT + n0 + n];
  }
  __syncthreads();
  for (int e = threadIdx.x; e < 64 * 64; e += 256) {
    int n = e >> 6, g = e & 63;
    wN[((size_t)b * NT + n0 + n) * INNER + h * G + g] = T[g][n];
  }
}

// ---------------- conv 3x3 as implicit GEMM (bf16 WMMA, LDS-free) ------------
// Block = 4 waves; block tile M=16 pixels x N=256; wave tile 16x64 (4 accumulators).
__global__ void __launch_bounds__(128)
conv3x3_wmma(const bf16* __restrict__ xb, const bf16* __restrict__ wBT,
             const float* __restrict__ convB, float* __restrict__ xp) {
  using namespace cfg;
  int nt = blockIdx.x & 1;
  int mt = (blockIdx.x >> 1) % (NT / 16);
  int bi = blockIdx.x / (2 * (NT / 16));
  int tid = threadIdx.x, lane = tid & 31, wv = tid >> 5;
  int m = lane & 15, hf = (lane >> 4) & 1;
  int px = mt * 16 + m;                 // 16 | 160 -> tile never crosses an image row
  int y0 = px / Ww, x0 = px % Ww;
  v8f zero8 = {0, 0, 0, 0, 0, 0, 0, 0};
  v8f acc[4] = {zero8, zero8, zero8, zero8};
  v16bf za;
#pragma unroll
  for (int j = 0; j < 16; ++j) za[j] = (bf16)0.f;
  int nb = nt * 256 + wv * 64;
  const bf16* bcol[4];
#pragma unroll
  for (int j = 0; j < 4; ++j) bcol[j] = wBT + (size_t)(nb + j * 16 + m) * KC;
  for (int ks = 0; ks < KC / 32; ++ks) {
    int kb = ks * 32;
    int plane = kb >> 8;                 // 32 | 256 -> one (ky,kx) tap per k-step
    int ky = plane / 3 - 1, kx = plane % 3 - 1;
    int cin = kb & 255;
    int yy = y0 + ky, xx = x0 + kx;
    v16bf a = za;
    if (yy >= 0 && yy < Hh && xx >= 0 && xx < Ww) {
      const bf16* ap = xb + (((size_t)bi * Hh + yy) * Ww + xx) * DIM + cin;
      a = frag16(ap + hf * 8, ap + 16 + hf * 8);
      __builtin_prefetch(ap + 32, 0, 3);   // next k-step activations (global_prefetch_b8)
    }
#pragma unroll
    for (int j = 0; j < 4; ++j) {
      v16bf b = frag16(bcol[j] + kb + hf * 8, bcol[j] + kb + 16 + hf * 8);
      acc[j] = __builtin_amdgcn_wmma_f32_16x16x32_bf16(false, a, false, b, (short)0, acc[j],
                                                       false, false);
    }
  }
  int rh = hf * 8;
#pragma unroll
  for (int j = 0; j < 4; ++j) {
    int col = nb + j * 16 + m;
    float bias = convB[col];
#pragma unroll
    for (int r = 0; r < 8; ++r) {
      int p = mt * 16 + r + rh;
      xp[((size_t)bi * NT + p) * INNER + col] = acc[j][r] + bias;
    }
  }
}

// ---------------- slice logits + temperature ---------------------------------
__global__ void __launch_bounds__(64)
logits_kernel(const float* __restrict__ xp, const float* __restrict__ sw,
              const float* __restrict__ sb, const float* __restrict__ aw,
              const float* __restrict__ ab, float* __restrict__ zT, float lnle) {
  using namespace cfg;
  int n = blockIdx.x % NT;
  int bh = blockIdx.x / NT;
  int b = bh >> 3, h = bh & 7;
  __shared__ float row[64];
  __shared__ float sada;
  int t = threadIdx.x;
  // Stage the token row via the CDNA5 async global->LDS path (ASYNCcnt-tracked).
  {
    unsigned ldsoff = (unsigned)(unsigned long long)&row[t];
    unsigned long long ga =
        (unsigned long long)(xp + ((size_t)b * NT + n) * INNER + h * DH + t);
    asm volatile("global_load_async_to_lds_b32 %0, %1, off" ::"v"(ldsoff), "v"(ga) : "memory");
    asm volatile("s_wait_asynccnt 0x0" ::: "memory");
  }
  __syncthreads();
  if (t == 0) {
    float s = ab[0];
    for (int c = 0; c < DH; ++c) s += row[c] * aw[c];
    s = fminf(0.4f, fmaxf(-0.4f, s));
    sada = 0.5f + s;
  }
  __syncthreads();
  float d = sb[t] - lnle;
  const float* w = sw + t * DH;
  for (int c = 0; c < DH; ++c) d += row[c] * w[c];
  zT[((size_t)bh * G + t) * NT + n] = d / sada;
}

__global__ void __launch_bounds__(256)
softmax_reduce(const float* __restrict__ zT, float* __restrict__ cmax, float* __restrict__ csum) {
  using namespace cfg;
  int col = blockIdx.x;
  const float* z = zT + (size_t)col * NT;
  __shared__ float red[256];
  int t = threadIdx.x;
  float m = -3.4e38f;
  for (int i = t; i < NT; i += 256) m = fmaxf(m, z[i]);
  red[t] = m; __syncthreads();
  for (int s = 128; s > 0; s >>= 1) { if (t < s) red[t] = fmaxf(red[t], red[t + s]); __syncthreads(); }
  float M = red[0]; __syncthreads();
  float sa = 0.f;
  for (int i = t; i < NT; i += 256) sa += expf(z[i] - M);
  red[t] = sa; __syncthreads();
  for (int s = 128; s > 0; s >>= 1) { if (t < s) red[t] += red[t + s]; __syncthreads(); }
  if (t == 0) { cmax[col] = M; csum[col] = red[0]; }
}

__global__ void softmax_norm(const float* __restrict__ zT, const float* __restrict__ cmax,
                             const float* __restrict__ csum, bf16* __restrict__ wT) {
  using namespace cfg;
  size_t n = (size_t)NBH * G * NT;
  for (size_t i = blockIdx.x * (size_t)blockDim.x + threadIdx.x; i < n;
       i += (size_t)gridDim.x * blockDim.x) {
    size_t col = i / NT;
    wT[i] = (bf16)(expf(zT[i] - cmax[col]) / csum[col]);
  }
}

// ---------------- eid = wT * xproj^T (K = 25600, LDS-free WMMA) --------------
// Block = 4 waves; wave wv owns g rows [wv*16, wv*16+16), all 64 channels (4 acc).
__global__ void __launch_bounds__(128)
eid_wmma(const bf16* __restrict__ wT, const bf16* __restrict__ xpT, float* __restrict__ es) {
  using namespace cfg;
  int bh = blockIdx.x;
  int tid = threadIdx.x, lane = tid & 31, wv = tid >> 5;
  int m = lane & 15, hf = (lane >> 4) & 1;
  v8f zero8 = {0, 0, 0, 0, 0, 0, 0, 0};
  v8f acc[4] = {zero8, zero8, zero8, zero8};
  const bf16* arow = wT + ((size_t)bh * G + wv * 16 + m) * NT;
  const bf16* bcol[4];
#pragma unroll
  for (int j = 0; j < 4; ++j) bcol[j] = xpT + ((size_t)bh * DH + j * 16 + m) * NT;
  for (int ks = 0; ks < NT / 32; ++ks) {
    int kb = ks * 32;
    v16bf a = frag16(arow + kb + hf * 8, arow + kb + 16 + hf * 8);
#pragma unroll
    for (int j = 0; j < 4; ++j) {
      v16bf b = frag16(bcol[j] + kb + hf * 8, bcol[j] + kb + 16 + hf * 8);
      acc[j] = __builtin_amdgcn_wmma_f32_16x16x32_bf16(false, a, false, b, (short)0, acc[j],
                                                       false, false);
    }
  }
  int rh = hf * 8;
  const float inv = 1.f / (1.f + 1e-5f);  // softmax over n sums to 1
#pragma unroll
  for (int j = 0; j < 4; ++j)
#pragma unroll
    for (int r = 0; r < 8; ++r) {
      int g = wv * 16 + r + rh;
      es[((size_t)bh * G + g) * DH + j * 16 + m] = acc[j][r] * inv;
    }
}

// ---------------- geometry: feature min/max -> positions ---------------------
__global__ void __launch_bounds__(64)
minmax_pos(const float* __restrict__ es, float* __restrict__ pos) {
  using namespace cfg;
  __shared__ float smn[64], smx[64];
  __shared__ float gmn[2], gmx[2];
  int t = threadIdx.x;
  int c = t >> 5, lane = t & 31;
  float mn = 3.4e38f, mx = -3.4e38f;
  for (int r = lane; r < NBH * G; r += 32) {
    float v = es[(size_t)r * DH + c];
    mn = fminf(mn, v); mx = fmaxf(mx, v);
  }
  smn[t] = mn; smx[t] = mx; __syncthreads();
  if (lane == 0) {
    float a = smn[c * 32], b2 = smx[c * 32];
    for (int k = 1; k < 32; ++k) { a = fminf(a, smn[c * 32 + k]); b2 = fmaxf(b2, smx[c * 32 + k]); }
    gmn[c] = a; gmx[c] = b2;
  }
  __syncthreads();
  for (int e = t; e < NBH * G * 2; e += 64) {
    int r = e >> 1, cc = e & 1;
    pos[e] = (es[(size_t)r * DH + cc] - gmn[cc]) / (gmx[cc] - gmn[cc] + 1e-8f);
  }
}

// ---------------- balanced ball-tree permutation -----------------------------
__global__ void __launch_bounds__(64)
balltree_kernel(const float* __restrict__ es, const float* __restrict__ pos,
                float* __restrict__ xe, float* __restrict__ posp,
                int* __restrict__ perm, int* __restrict__ invp) {
  using namespace cfg;
  int seq = blockIdx.x, t = threadIdx.x;
  __shared__ int idx[64];
  __shared__ float px[64], py[64];
  idx[t] = t;
  px[t] = pos[(size_t)(seq * G + t) * 2];
  py[t] = pos[(size_t)(seq * G + t) * 2 + 1];
  __syncthreads();
  if (t == 0) {
    for (int seg = G; seg > 2; seg >>= 1) {
      int ns = G / seg;
      for (int s = 0; s < ns; ++s) {
        int off = s * seg;
        float mnx = 3e38f, mxx = -3e38f, mny = 3e38f, mxy = -3e38f;
        for (int i = 0; i < seg; ++i) {
          int id = idx[off + i];
          mnx = fminf(mnx, px[id]); mxx = fmaxf(mxx, px[id]);
          mny = fminf(mny, py[id]); mxy = fmaxf(mxy, py[id]);
        }
        int axis = ((mxx - mnx) >= (mxy - mny)) ? 0 : 1;
        for (int i = 1; i < seg; ++i) {  // stable insertion sort
          int id = idx[off + i];
          float key = axis ? py[id] : px[id];
          int j = i - 1;
          while (j >= 0) {
            int jd = idx[off + j];
            float kj = axis ? py[jd] : px[jd];
            if (kj > key) { idx[off + j + 1] = jd; --j; } else break;
          }
          idx[off + j + 1] = id;
        }
      }
    }
  }
  __syncthreads();
  int p = idx[t];
  perm[seq * G + t] = p;
  invp[seq * G + p] = t;
  for (int c = 0; c < DH; ++c)
    xe[(size_t)(seq * G + t) * DH + c] = es[(size_t)(seq * G + p) * DH + c];
  posp[(size_t)(seq * G + t) * 2] = px[p];
  posp[(size_t)(seq * G + t) * 2 + 1] = py[p];
}

// ---------------- small generic kernels (Erwin) ------------------------------
__global__ void __launch_bounds__(64)
ln_kernel(const float* __restrict__ X, const float* __restrict__ g,
          const float* __restrict__ b, float* __restrict__ Y, int dim) {
  int r = blockIdx.x, t = threadIdx.x;
  __shared__ float s1[64], s2[64];
  float sum = 0.f, sq = 0.f;
  for (int i = t; i < dim; i += 64) { float v = X[(size_t)r * dim + i]; sum += v; sq += v * v; }
  s1[t] = sum; s2[t] = sq; __syncthreads();
  for (int s = 32; s > 0; s >>= 1) {
    if (t < s) { s1[t] += s1[t + s]; s2[t] += s2[t + s]; }
    __syncthreads();
  }
  float mean = s1[0] / dim;
  float var = s2[0] / dim - mean * mean;
  float inv = rsqrtf(var + 1e-5f);
  for (int i = t; i < dim; i += 64) {
    float v = X[(size_t)r * dim + i];
    Y[(size_t)r * dim + i] = (v - mean) * inv * g[i] + b[i];
  }
}

__global__ void __launch_bounds__(128)
linear_kernel(const float* __restrict__ X, const float* __restrict__ W,
              const float* __restrict__ bias, const float* __restrict__ res,
              float* __restrict__ Y, int inDim, int outDim) {
  int r = blockIdx.x;
  __shared__ float row[512];
  for (int i = threadIdx.x; i < inDim; i += 128) row[i] = X[(size_t)r * inDim + i];
  __syncthreads();
  for (int o = threadIdx.x; o < outDim; o += 128) {
    float s = bias ? bias[o] : 0.f;
    const float* w = W + (size_t)o * inDim;
    for (int i = 0; i < inDim; ++i) s += row[i] * w[i];
    if (res) s += res[(size_t)r * outDim + o];
    Y[(size_t)r * outDim + o] = s;
  }
}

__global__ void gate_kernel(float* __restrict__ A, const float* __restrict__ Bv, int n) {
  for (int i = blockIdx.x * blockDim.x + threadIdx.x; i < n; i += gridDim.x * blockDim.x) {
    float x = A[i];
    A[i] = (x / (1.f + expf(-x))) * Bv[i];
  }
}

__global__ void copy_kernel(const float* __restrict__ a, float* __restrict__ b, size_t n) {
  for (size_t i = blockIdx.x * (size_t)blockDim.x + threadIdx.x; i < n;
       i += (size_t)gridDim.x * blockDim.x)
    b[i] = a[i];
}

// ---------------- ball attention (one block per ball) ------------------------
__global__ void __launch_bounds__(128)
ball_msa_kernel(const float* __restrict__ qkv_w, const float* __restrict__ qkv_b,
                const float* __restrict__ proj_w, const float* __restrict__ proj_b,
                const float* __restrict__ pe_w, const float* __restrict__ pe_b,
                const float* __restrict__ sigma,
                const float* __restrict__ Hin, const float* __restrict__ POS,
                float* __restrict__ A,
                int nSeq, int dim, int heads, int ball, int shift) {
  __shared__ float XB[2048];
  __shared__ float PB[64];
  __shared__ float QKV[6144];
  __shared__ float ATT[4096];
  __shared__ float OB[2048];
  __shared__ float cm[2];
  int bps = nSeq / ball;
  int seq = blockIdx.x / bps;
  int bl = blockIdx.x % bps;
  int t = threadIdx.x;
  int tdim = ball * dim;
  for (int e = t; e < tdim; e += 128) {
    int i = e / dim, c = e % dim;
    int src = seq * nSeq + ((bl * ball + i - shift + nSeq) % nSeq);  // roll by -shift
    XB[e] = Hin[(size_t)src * dim + c];
  }
  for (int e = t; e < ball * 2; e += 128) {
    int i = e >> 1, c = e & 1;
    int src = seq * nSeq + ((bl * ball + i - shift + nSeq) % nSeq);
    PB[e] = POS[(size_t)src * 2 + c];
  }
  __syncthreads();
  if (t < 2) {
    float s = 0.f;
    for (int i = 0; i < ball; ++i) s += PB[i * 2 + t];
    cm[t] = s / ball;
  }
  __syncthreads();
  for (int e = t; e < tdim; e += 128) {
    int i = e / dim, c = e % dim;
    float rx = PB[i * 2] - cm[0], ry = PB[i * 2 + 1] - cm[1];
    XB[e] += rx * pe_w[c * 2] + ry * pe_w[c * 2 + 1] + pe_b[c];
  }
  __syncthreads();
  int d3 = 3 * dim;
  for (int e = t; e < ball * d3; e += 128) {
    int i = e / d3, j = e % d3;
    float s = qkv_b[j];
    const float* w = qkv_w + (size_t)j * dim;
    const float* xr = XB + i * dim;
    for (int c = 0; c < dim; ++c) s += xr[c] * w[c];
    QKV[e] = s;
  }
  __syncthreads();
  int hd = dim / heads;
  float scale = rsqrtf((float)hd);
  for (int row = t; row < heads * ball; row += 128) {
    int hh = row / ball, i = row % ball;
    const float* q = QKV + (size_t)i * d3 + hh * hd;
    float pix = PB[i * 2], piy = PB[i * 2 + 1];
    float sg = sigma[hh];
    float mx = -3.4e38f;
    for (int j = 0; j < ball; ++j) {
      const float* k = QKV + (size_t)j * d3 + dim + hh * hd;
      float dt = 0.f;
      for (int c = 0; c < hd; ++c) dt += q[c] * k[c];
      float dx = pix - PB[j * 2], dy = piy - PB[j * 2 + 1];
      float l = dt * scale + sg * sqrtf(dx * dx + dy * dy + 1e-12f);
      ATT[row * ball + j] = l;
      mx = fmaxf(mx, l);
    }
    float ss = 0.f;
    for (int j = 0; j < ball; ++j) { float e2 = expf(ATT[row * ball + j] - mx); ATT[row * ball + j] = e2; ss += e2; }
    float inv = 1.f / ss;
    for (int j = 0; j < ball; ++j) ATT[row * ball + j] *= inv;
  }
  __syncthreads();
  for (int e = t; e < tdim; e += 128) {
    int i = e / dim, c = e % dim, hh = c / hd;
    float s = 0.f;
    for (int j = 0; j < ball; ++j)
      s += ATT[(hh * ball + i) * ball + j] * QKV[(size_t)j * d3 + 2 * dim + c];
    OB[e] = s;
  }
  __syncthreads();
  for (int e = t; e < tdim; e += 128) {
    int i = e / dim, c = e % dim;
    float s = proj_b[c];
    const float* w = proj_w + (size_t)c * dim;
    const float* orow = OB + i * dim;
    for (int c2 = 0; c2 < dim; ++c2) s += orow[c2] * w[c2];
    int src = seq * nSeq + ((bl * ball + i - shift + nSeq) % nSeq);
    A[(size_t)src * dim + c] += s;
  }
}

// ---------------- pool / unpool prep -----------------------------------------
__global__ void __launch_bounds__(64)
pool_prep(const float* __restrict__ A0, const float* __restrict__ POSP,
          float* __restrict__ CAT, float* __restrict__ POS1) {
  int r = blockIdx.x, t = threadIdx.x;
  CAT[(size_t)r * 132 + t] = A0[(size_t)(2 * r) * 64 + t];
  CAT[(size_t)r * 132 + 64 + t] = A0[(size_t)(2 * r + 1) * 64 + t];
  if (t == 0) {
    float p0x = POSP[(size_t)(2 * r) * 2], p0y = POSP[(size_t)(2 * r) * 2 + 1];
    float p1x = POSP[(size_t)(2 * r + 1) * 2], p1y = POSP[(size_t)(2 * r + 1) * 2 + 1];
    float cx = 0.5f * (p0x + p1x), cy = 0.5f * (p0y + p1y);
    POS1[r * 2] = cx; POS1[r * 2 + 1] = cy;
    CAT[(size_t)r * 132 + 128] = p0x - cx; CAT[(size_t)r * 132 + 129] = p0y - cy;
    CAT[(size_t)r * 132 + 130] = p1x - cx; CAT[(size_t)r * 132 + 131] = p1y - cy;
  }
}

__global__ void __launch_bounds__(128)
unpool_prep(const float* __restrict__ A1, const float* __restrict__ POSP, float* __restrict__ CAT) {
  int r = blockIdx.x, t = threadIdx.x;
  CAT[(size_t)r * 132 + t] = A1[(size_t)r * 128 + t];
  if (t == 0) {
    float p0x = POSP[(size_t)(2 * r) * 2], p0y = POSP[(size_t)(2 * r) * 2 + 1];
    float p1x = POSP[(size_t)(2 * r + 1) * 2], p1y = POSP[(size_t)(2 * r + 1) * 2 + 1];
    float cx = 0.5f * (p0x + p1x), cy = 0.5f * (p0y + p1y);
    CAT[(size_t)r * 132 + 128] = p0x - cx; CAT[(size_t)r * 132 + 129] = p0y - cy;
    CAT[(size_t)r * 132 + 130] = p1x - cx; CAT[(size_t)r * 132 + 131] = p1y - cy;
  }
}

__global__ void __launch_bounds__(64)
unperm_kernel(const float* __restrict__ A0, const int* __restrict__ invp, float* __restrict__ ep) {
  int seq = blockIdx.x, t = threadIdx.x;
  int src = invp[seq * 64 + t];
  for (int c = 0; c < 64; ++c)
    ep[(size_t)(seq * 64 + t) * 64 + c] = A0[(size_t)(seq * 64 + src) * 64 + c];
}

// ---------------- fused de-slice: MT[b][o][h*64+g] = e . out_w ---------------
__global__ void __launch_bounds__(256)
m_kernel(const float* __restrict__ ep, const float* __restrict__ outw, bf16* __restrict__ MT) {
  using namespace cfg;
  int k = blockIdx.x % 512;
  int b = blockIdx.x / 512;
  int h = k >> 6, g = k & 63;
  int bh = b * 8 + h;
  __shared__ float e[64];
  int t = threadIdx.x;
  if (t < 64) e[t] = ep[(size_t)(bh * 64 + g) * 64 + t];
  __syncthreads();
  float s = 0.f;
  const float* w = outw + (size_t)t * INNER + h * 64;
  for (int c = 0; c < 64; ++c) s += e[c] * w[c];
  MT[((size_t)b * DIM + t) * INNER + k] = (bf16)s;  // K-contiguous per output column
}

// ---------------- out = wN(n,k) * MT(k,o) + bias (K=512, LDS-free WMMA) ------
__global__ void __launch_bounds__(128)
final_wmma(const bf16* __restrict__ wN, const bf16* __restrict__ MT,
           const float* __restrict__ outb, float* __restrict__ out) {
  using namespace cfg;
  int mt = blockIdx.x % (NT / 16);
  int b = blockIdx.x / (NT / 16);
  int tid = threadIdx.x, lane = tid & 31, wv = tid >> 5;
  int m = lane & 15, hf = (lane >> 4) & 1;
  v8f zero8 = {0, 0, 0, 0, 0, 0, 0, 0};
  v8f acc[4] = {zero8, zero8, zero8, zero8};
  const bf16* arow = wN + ((size_t)b * NT + mt * 16 + m) * INNER;
  const bf16* bcol[4];
#pragma unroll
  for (int j = 0; j < 4; ++j) bcol[j] = MT + ((size_t)b * DIM + wv * 64 + j * 16 + m) * INNER;
  for (int ks = 0; ks < INNER / 32; ++ks) {
    int kb = ks * 32;
    v16bf a = frag16(arow + kb + hf * 8, arow + kb + 16 + hf * 8);
#pragma unroll
    for (int j = 0; j < 4; ++j) {
      v16bf bb = frag16(bcol[j] + kb + hf * 8, bcol[j] + kb + 16 + hf * 8);
      acc[j] = __builtin_amdgcn_wmma_f32_16x16x32_bf16(false, a, false, bb, (short)0, acc[j],
                                                       false, false);
    }
  }
  int rh = hf * 8;
#pragma unroll
  for (int j = 0; j < 4; ++j) {
    int o = wv * 64 + j * 16 + m;
    float bias = outb[o];
#pragma unroll
    for (int r = 0; r < 8; ++r) {
      int n = mt * 16 + r + rh;
      out[((size_t)b * NT + n) * DIM + o] = acc[j][r] + bias;
    }
  }
}

// ---------------- host orchestration -----------------------------------------
struct BlockP {
  const float *ln1_s, *ln1_b, *ln2_s, *ln2_b, *qkv_w, *qkv_b, *proj_w, *proj_b,
      *pe_w, *pe_b, *sigma, *w1, *b1, *w2, *b2, *w3, *b3;
};

static BlockP get_block(void* const* d_in, int base) {
  auto F = [&](int i) { return (const float*)d_in[i]; };
  BlockP p;
  p.ln1_s = F(base + 0); p.ln1_b = F(base + 1); p.ln2_s = F(base + 2); p.ln2_b = F(base + 3);
  p.qkv_w = F(base + 4); p.qkv_b = F(base + 5); p.proj_w = F(base + 6); p.proj_b = F(base + 7);
  p.pe_w = F(base + 8); p.pe_b = F(base + 9); p.sigma = F(base + 10);
  p.w1 = F(base + 11); p.b1 = F(base + 12); p.w2 = F(base + 13); p.b2 = F(base + 14);
  p.w3 = F(base + 15); p.b3 = F(base + 16);
  return p;
}

static void run_block(hipStream_t st, const BlockP& p, float* A, float* Hb, float* Ma, float* Mb,
                      const float* POS, int rows, int nSeq, int dim, int heads, int ball, bool rot) {
  ln_kernel<<<rows, 64, 0, st>>>(A, p.ln1_s, p.ln1_b, Hb, dim);
  ball_msa_kernel<<<rows / ball, 128, 0, st>>>(p.qkv_w, p.qkv_b, p.proj_w, p.proj_b, p.pe_w,
                                               p.pe_b, p.sigma, Hb, POS, A, nSeq, dim, heads,
                                               ball, rot ? ball / 2 : 0);
  ln_kernel<<<rows, 64, 0, st>>>(A, p.ln2_s, p.ln2_b, Hb, dim);
  int hh = 4 * dim;
  linear_kernel<<<rows, 128, 0, st>>>(Hb, p.w1, p.b1, nullptr, Ma, dim, hh);
  linear_kernel<<<rows, 128, 0, st>>>(Hb, p.w2, p.b2, nullptr, Mb, dim, hh);
  int n = rows * hh;
  gate_kernel<<<(n + 255) / 256, 256, 0, st>>>(Ma, Mb, n);
  linear_kernel<<<rows, 128, 0, st>>>(Ma, p.w3, p.b3, A, A, hh, dim);
}

extern "C" void kernel_launch(void* const* d_in, const int* in_sizes, int n_in,
                              void* d_out, int out_size, void* d_ws, size_t ws_size,
                              hipStream_t stream) {
  using namespace cfg;
  (void)in_sizes; (void)n_in; (void)out_size; (void)ws_size;
  auto F = [&](int i) { return (const float*)d_in[i]; };
  const float* x = F(0);
  const float* conv_w = F(1); const float* conv_b = F(2);
  const float* slice_w = F(3); const float* slice_b = F(4);
  const float* ada_w = F(5); const float* ada_b = F(6);
  const float* out_w = F(7); const float* out_b = F(8);
  const float* embed_w = F(9); const float* embed_b = F(10);
  BlockP enc0_0 = get_block(d_in, 11), enc0_1 = get_block(d_in, 28);
  const float* pool_w = F(45); const float* pool_b = F(46);
  BlockP enc1_0 = get_block(d_in, 47), enc1_1 = get_block(d_in, 64);
  const float* unpool_w = F(81); const float* unpool_b = F(82);
  BlockP dec0_0 = get_block(d_in, 83), dec0_1 = get_block(d_in, 100);

  char* base = (char*)d_ws; size_t off = 0;
  auto alloc = [&](size_t bytes) { void* p = base + off; off = (off + bytes + 255) & ~(size_t)255; return p; };
  bf16*  xb   = (bf16*)alloc((size_t)B * NT * DIM * 2);
  bf16*  wBT  = (bf16*)alloc((size_t)KC * INNER * 2);
  float* xp   = (float*)alloc((size_t)B * NT * INNER * 4);
  bf16*  xpT  = (bf16*)alloc((size_t)NBH * NT * DH * 2);
  float* zT   = (float*)alloc((size_t)NBH * G * NT * 4);
  bf16*  wT   = (bf16*)alloc((size_t)NBH * G * NT * 2);
  bf16*  wN   = (bf16*)alloc((size_t)B * NT * INNER * 2);
  float* cmax = (float*)alloc((size_t)NBH * G * 4);
  float* csum = (float*)alloc((size_t)NBH * G * 4);
  float* es   = (float*)alloc((size_t)NBH * G * DH * 4);
  float* posb = (float*)alloc((size_t)NBH * G * 2 * 4);
  int*   perm = (int*)alloc((size_t)NBH * G * 4);
  int*   invp = (int*)alloc((size_t)NBH * G * 4);
  float* xe   = (float*)alloc((size_t)NBH * G * DH * 4);
  float* posp = (float*)alloc((size_t)NBH * G * 2 * 4);
  float* A0   = (float*)alloc((size_t)NBH * G * DH * 4);     // also viewed [512,128]
  float* H0   = (float*)alloc((size_t)NBH * G * DH * 4);
  float* A1   = (float*)alloc((size_t)NBH * 32 * 128 * 4);
  float* H1   = (float*)alloc((size_t)NBH * 32 * 128 * 4);
  float* Ma   = (float*)alloc((size_t)NBH * G * 256 * 4);
  float* Mb   = (float*)alloc((size_t)NBH * G * 256 * 4);
  float* SK   = (float*)alloc((size_t)NBH * G * DH * 4);
  float* POS1 = (float*)alloc((size_t)NBH * 32 * 2 * 4);
  float* CAT  = (float*)alloc((size_t)NBH * 32 * 132 * 4);
  float* ep   = (float*)alloc((size_t)NBH * G * DH * 4);
  bf16*  MT   = (bf16*)alloc((size_t)B * DIM * INNER * 2);

  const float lnle = logf(-logf(1e-6f));

  cvt_f32_bf16<<<4096, 256, 0, stream>>>(x, xb, (size_t)B * NT * DIM);
  cvt_convw<<<2048, 256, 0, stream>>>(conv_w, wBT);
  conv3x3_wmma<<<B * (NT / 16) * 2, 128, 0, stream>>>(xb, wBT, conv_b, xp);
  transpose_xproj<<<NBH * (NT / 64), 256, 0, stream>>>(xp, xpT);
  logits_kernel<<<NBH * NT, 64, 0, stream>>>(xp, slice_w, slice_b, ada_w, ada_b, zT, lnle);
  softmax_reduce<<<NBH * G, 256, 0, stream>>>(zT, cmax, csum);
  softmax_norm<<<8192, 256, 0, stream>>>(zT, cmax, csum, wT);
  transpose_wn<<<NBH * (NT / 64), 256, 0, stream>>>(wT, wN);
  eid_wmma<<<NBH, 128, 0, stream>>>(wT, xpT, es);
  minmax_pos<<<1, 64, 0, stream>>>(es, posb);
  balltree_kernel<<<NBH, 64, 0, stream>>>(es, posb, xe, posp, perm, invp);

  // Erwin
  linear_kernel<<<NBH * G, 128, 0, stream>>>(xe, embed_w, embed_b, nullptr, A0, DH, DH);
  run_block(stream, enc0_0, A0, H0, Ma, Mb, posp, NBH * G, G, 64, 4, 32, false);
  run_block(stream, enc0_1, A0, H0, Ma, Mb, posp, NBH * G, G, 64, 4, 32, true);
  copy_kernel<<<256, 256, 0, stream>>>(A0, SK, (size_t)NBH * G * DH);
  pool_prep<<<NBH * 32, 64, 0, stream>>>(A0, posp, CAT, POS1);
  linear_kernel<<<NBH * 32, 128, 0, stream>>>(CAT, pool_w, pool_b, nullptr, A1, 132, 128);
  run_block(stream, enc1_0, A1, H1, Ma, Mb, POS1, NBH * 32, 32, 128, 8, 16, false);
  run_block(stream, enc1_1, A1, H1, Ma, Mb, POS1, NBH * 32, 32, 128, 8, 16, true);
  unpool_prep<<<NBH * 32, 128, 0, stream>>>(A1, posp, CAT);
  linear_kernel<<<NBH * 32, 128, 0, stream>>>(CAT, unpool_w, unpool_b, SK, A0, 132, 128);
  run_block(stream, dec0_0, A0, H0, Ma, Mb, posp, NBH * G, G, 64, 4, 32, false);
  run_block(stream, dec0_1, A0, H0, Ma, Mb, posp, NBH * G, G, 64, 4, 32, true);
  unperm_kernel<<<NBH, 64, 0, stream>>>(A0, invp, ep);

  // fused de-slice + output projection
  m_kernel<<<B * 512, 256, 0, stream>>>(ep, out_w, MT);
  final_wmma<<<B * (NT / 16), 128, 0, stream>>>(wN, MT, out_b, (float*)d_out);
}
